// TravelDLRM_5403068858959
// MI455X (gfx1250) — matrix operand order repaired and checked by reading, not compiled, not run on hardware
//
#include <hip/hip_runtime.h>
#include <hip/hip_bf16.h>
#include <math.h>

typedef __bf16 v16bf __attribute__((ext_vector_type(16)));
typedef __bf16 v8bf  __attribute__((ext_vector_type(8)));
typedef float  v8f   __attribute__((ext_vector_type(8)));

#define BM 128
#define BN 128
#define BK 32
#define LDA 40   // bf16 elems per LDS row (80B = 5*16B: aligned + conflict-friendly)
#define LDB 40

static __device__ inline v16bf cat8(v8bf lo, v8bf hi)
{
    return __builtin_shufflevector(lo, hi, 0, 1, 2, 3, 4, 5, 6, 7,
                                   8, 9, 10, 11, 12, 13, 14, 15);
}

// Async direct global->LDS copy, 16B per lane (ASYNCcnt-tracked).
// lds_off: wave-relative LDS byte offset (low 32 bits of flat shared pointer).
static __device__ inline void async_copy16(unsigned lds_off, const void* gaddr)
{
    asm volatile("global_load_async_to_lds_b128 %0, %1, off"
                 :: "v"(lds_off),
                    "v"((unsigned long long)(uintptr_t)gaddr)
                 : "memory");
}

static __device__ inline void wait_async0()
{
    asm volatile("s_wait_asynccnt 0x0" ::: "memory");
}

// ---------------------------------------------------------------------------
// Generic bf16 WMMA GEMM:  Y[m][n] = act( (X @ W^T)[m][n] * scale[n] + shift[n] )
// X: M x K (row-major, bf16), W: N x K (row-major, bf16), K % 32 == 0, M % 128 == 0.
// act: 0 = none, 1 = relu, 2 = tanh. Writes bf16 (always) and f32 (optional).
// Block tile 128x128, 8 waves; each wave computes a 32x64 strip (2x4 WMMA tiles).
// Tiles are staged with GLOBAL_LOAD_ASYNC_TO_LDS_B128 (no VGPR staging).
// Rows of W past N are clamped (not zeroed): they only feed output columns
// n >= N, which the epilogue never stores.
// ---------------------------------------------------------------------------
__global__ __launch_bounds__(256) void k_gemm_bf16(
    const __bf16* __restrict__ X, const __bf16* __restrict__ W,
    const float* __restrict__ scale, const float* __restrict__ shift,
    __bf16* __restrict__ Yb, int ldYb,
    float* __restrict__ Yf, int ldYf,
    int M, int N, int K, int act)
{
    __shared__ __align__(16) __bf16 As[BM * LDA];
    __shared__ __align__(16) __bf16 Bs[BN * LDB];

    const int tid  = threadIdx.x;
    const int lane = tid & 31;
    const int wave = tid >> 5;
    const int wm   = wave & 3;      // 4 M-strips of 32 rows
    const int wn   = wave >> 2;     // 2 N-strips of 64 cols
    const int m0   = blockIdx.x * BM;
    const int n0   = blockIdx.y * BN;
    const int r    = lane & 15;
    const int kh   = lane >> 4;

    // cooperative-load coordinates: 2 threads per row, 16 bf16 (32B) each
    const int crow = tid >> 1;
    const int ccol = (tid & 1) * 16;
    int nrow = n0 + crow;
    if (nrow > N - 1) nrow = N - 1;   // clamp: garbage cols are never stored

    const unsigned ldsA0 = (unsigned)(uintptr_t)&As[crow * LDA + ccol];
    const unsigned ldsA1 = (unsigned)(uintptr_t)&As[crow * LDA + ccol + 8];
    const unsigned ldsB0 = (unsigned)(uintptr_t)&Bs[crow * LDB + ccol];
    const unsigned ldsB1 = (unsigned)(uintptr_t)&Bs[crow * LDB + ccol + 8];
    const __bf16* gA = X + (size_t)(m0 + crow) * K + ccol;
    const __bf16* gB = W + (size_t)nrow * K + ccol;

    v8f acc[2][4] = {};

    for (int k0 = 0; k0 < K; k0 += BK) {
        __syncthreads();
        async_copy16(ldsA0, gA + k0);
        async_copy16(ldsA1, gA + k0 + 8);
        async_copy16(ldsB0, gB + k0);
        async_copy16(ldsB1, gB + k0 + 8);
        wait_async0();
        __syncthreads();

        // A fragments: lanes 0-15 -> K 0..7 / 16..23 ; lanes 16-31 -> K 8..15 / 24..31
        v16bf a[2], b[4];
        #pragma unroll
        for (int mi = 0; mi < 2; ++mi) {
            const __bf16* rp = &As[(wm * 32 + mi * 16 + r) * LDA];
            v8bf lo = *reinterpret_cast<const v8bf*>(rp + kh * 8);
            v8bf hi = *reinterpret_cast<const v8bf*>(rp + 16 + kh * 8);
            a[mi] = cat8(lo, hi);
        }
        // B fragments: col = lane&15, lanes 0-15 -> K 0..7 / 16..23 ; lanes 16-31 -> K 8..15 / 24..31
        #pragma unroll
        for (int ni = 0; ni < 4; ++ni) {
            const __bf16* rp = &Bs[(wn * 64 + ni * 16 + r) * LDB + kh * 8];
            v8bf lo = *reinterpret_cast<const v8bf*>(rp);
            v8bf hi = *reinterpret_cast<const v8bf*>(rp + 16);
            b[ni] = cat8(lo, hi);
        }
        #pragma unroll
        for (int mi = 0; mi < 2; ++mi)
            #pragma unroll
            for (int ni = 0; ni < 4; ++ni)
                acc[mi][ni] = __builtin_amdgcn_wmma_f32_16x16x32_bf16(
                    false, a[mi], false, b[ni], (short)0, acc[mi][ni], false, false);
    }

    // Epilogue: C/D layout  n = lane&15 (+16 per N subtile), m = (lane>>4)*8 + rr
    #pragma unroll
    for (int mi = 0; mi < 2; ++mi) {
        #pragma unroll
        for (int ni = 0; ni < 4; ++ni) {
            int n = n0 + wn * 64 + ni * 16 + r;
            if (n >= N) continue;
            float s = scale[n], sh = shift[n];
            #pragma unroll
            for (int rr = 0; rr < 8; ++rr) {
                int m = m0 + wm * 32 + mi * 16 + kh * 8 + rr;
                float v = acc[mi][ni][rr] * s + sh;
                if (act == 1) v = v > 0.0f ? v : 0.0f;
                else if (act == 2) v = tanhf(v);
                Yb[(size_t)m * ldYb + n] = (__bf16)v;
                if (Yf) Yf[(size_t)m * ldYf + n] = v;
            }
        }
    }
}

// ---------------------------------------------------------------------------
// Small helper kernels
// ---------------------------------------------------------------------------
__global__ void k_f32_to_bf16(const float* __restrict__ s, __bf16* __restrict__ d, int n)
{
    int i = blockIdx.x * blockDim.x + threadIdx.x;
    if (i < n) d[i] = (__bf16)s[i];
}

// Convert W (N x K) f32 -> bf16 with K padded to Kp (zero fill)
__global__ void k_convert_pad(const float* __restrict__ s, __bf16* __restrict__ d,
                              int N, int K, int Kp)
{
    int i = blockIdx.x * blockDim.x + threadIdx.x;
    if (i >= N * Kp) return;
    int n = i / Kp, k = i - n * Kp;
    d[i] = (k < K) ? (__bf16)s[(size_t)n * K + k] : (__bf16)0.0f;
}

// Fold BN into scale/shift:  s = g*rsqrt(rv+eps), shift = (b-rm)*s + bt
__global__ void k_scale_shift(const float* __restrict__ b, const float* __restrict__ g,
                              const float* __restrict__ bt, const float* __restrict__ rm,
                              const float* __restrict__ rv,
                              float* __restrict__ sc, float* __restrict__ sh,
                              int N, int has_bn)
{
    int i = blockIdx.x * blockDim.x + threadIdx.x;
    if (i >= N) return;
    if (has_bn) {
        float s = g[i] * rsqrtf(rv[i] + 1e-5f);
        sc[i] = s;
        sh[i] = (b[i] - rm[i]) * s + bt[i];
    } else {
        sc[i] = 1.0f;
        sh[i] = b[i];
    }
}

// Embedding gather -> all_emb_bf features 3..10 (layout B x 11 x 128)
__global__ void k_gather(const float* __restrict__ utab, const float* __restrict__ ptab,
                         const int* __restrict__ ucat, const int* __restrict__ pcat,
                         __bf16* __restrict__ all_emb, int B, int vocab)
{
    int i = blockIdx.x * blockDim.x + threadIdx.x;  // over B*8*128
    if (i >= B * 1024) return;
    int b = i >> 10, rem = i & 1023, t = rem >> 7, d = rem & 127;
    const float* tab;
    int ti, row;
    if (t < 4) { ti = t;     row = ucat[b * 4 + ti]; tab = utab; }
    else       { ti = t - 4; row = pcat[b * 4 + ti]; tab = ptab; }
    float v = tab[((size_t)ti * vocab + row) * 128 + d];
    all_emb[(size_t)b * 1408 + (3 + t) * 128 + d] = (__bf16)v;
}

// logits[b*11+f] = ba2 + sum_j h[(b*11+f)*32 + j] * Wa2[j]
__global__ void k_logits(const __bf16* __restrict__ h, const float* __restrict__ Wa2,
                         const float* __restrict__ ba2, float* __restrict__ lg, int n)
{
    int i = blockIdx.x * blockDim.x + threadIdx.x;
    if (i >= n) return;
    const __bf16* hp = h + (size_t)i * 32;
    float acc = ba2[0];
    #pragma unroll
    for (int j = 0; j < 32; ++j) acc += (float)hp[j] * Wa2[j];
    lg[i] = acc;
}

// Softmax over 11 features -> attn output (f32, straight into d_out region)
__global__ void k_softmax11(const float* __restrict__ lg, float* __restrict__ attn, int B)
{
    int b = blockIdx.x * blockDim.x + threadIdx.x;
    if (b >= B) return;
    float l[11], mx = -1e30f;
    #pragma unroll
    for (int f = 0; f < 11; ++f) { l[f] = lg[b * 11 + f]; mx = fmaxf(mx, l[f]); }
    float s = 0.0f;
    #pragma unroll
    for (int f = 0; f < 11; ++f) { l[f] = expf(l[f] - mx); s += l[f]; }
    float inv = 1.0f / s;
    #pragma unroll
    for (int f = 0; f < 11; ++f) attn[b * 11 + f] = l[f] * inv;
}

// bottom_for_int[b][d] = (1/11) * sum_f attn[b][f] * all_emb[b][f][d]
__global__ void k_bfi(const float* __restrict__ attn, const __bf16* __restrict__ all_emb,
                      __bf16* __restrict__ bfi, int B)
{
    int i = blockIdx.x * blockDim.x + threadIdx.x;  // over B*128
    if (i >= B * 128) return;
    int b = i >> 7, d = i & 127;
    float acc = 0.0f;
    #pragma unroll
    for (int f = 0; f < 11; ++f)
        acc += attn[b * 11 + f] * (float)all_emb[(size_t)b * 1408 + f * 128 + d];
    bfi[i] = (__bf16)(acc * (1.0f / 11.0f));
}

// Copy flat all_emb (1408) into top_in cols 66..1473
__global__ void k_flat(const __bf16* __restrict__ all_emb, __bf16* __restrict__ top_in, int B)
{
    int i = blockIdx.x * blockDim.x + threadIdx.x;  // over B*1408
    if (i >= B * 1408) return;
    int b = i / 1408, c = i - b * 1408;
    top_in[(size_t)b * 1504 + 66 + c] = all_emb[i];
}

// 66 upper-triangular interactions of 12x128 all_e -> top_in cols 0..65, zero pad 1474..1503
__global__ __launch_bounds__(128) void k_inter(const __bf16* __restrict__ bfi,
                                               const __bf16* __restrict__ all_emb,
                                               __bf16* __restrict__ top_in, int B)
{
    __shared__ float e[12 * 130];
    int b = blockIdx.x;
    int t = threadIdx.x;
    for (int i = t; i < 12 * 128; i += 128) {
        int f = i >> 7, d = i & 127;
        float v = (f == 0) ? (float)bfi[(size_t)b * 128 + d]
                           : (float)all_emb[(size_t)b * 1408 + (f - 1) * 128 + d];
        e[f * 130 + d] = v;
    }
    __syncthreads();
    if (t < 66) {
        int p = t, fi = 0, cnt = 11;
        while (p >= cnt) { p -= cnt; ++fi; --cnt; }
        int fj = fi + 1 + p;
        float acc = 0.0f;
        for (int d = 0; d < 128; ++d) acc += e[fi * 130 + d] * e[fj * 130 + d];
        top_in[(size_t)b * 1504 + t] = (__bf16)acc;
    } else if (t < 96) {
        top_in[(size_t)b * 1504 + 1474 + (t - 66)] = (__bf16)0.0f;
    }
}

// scores[b] = bf + dot(x[b], wf) over 128
__global__ void k_final(const __bf16* __restrict__ x, const __bf16* __restrict__ w,
                        const float* __restrict__ bf, float* __restrict__ out, int B)
{
    int b = blockIdx.x * blockDim.x + threadIdx.x;
    if (b >= B) return;
    float acc = bf[0];
    const __bf16* xp = x + (size_t)b * 128;
    #pragma unroll
    for (int k = 0; k < 128; ++k) acc += (float)xp[k] * (float)w[k];
    out[b] = acc;
}

// ---------------------------------------------------------------------------
// Host-side orchestration
// ---------------------------------------------------------------------------
extern "C" void kernel_launch(void* const* d_in, const int* in_sizes, int n_in,
                              void* d_out, int out_size, void* d_ws, size_t ws_size,
                              hipStream_t stream)
{
    (void)n_in; (void)out_size; (void)ws_size;
    const int B     = in_sizes[0] / 64;           // 32768
    const int vocab = in_sizes[3] / (4 * 128);    // 100000
    float* fo = (float*)d_out;

    // workspace bump allocator (256B aligned)
    char* ws = (char*)d_ws;
    size_t off = 0;
    auto allocB = [&](size_t n) -> __bf16* {
        off = (off + 255) & ~(size_t)255;
        __bf16* p = (__bf16*)(ws + off);
        off += n * sizeof(__bf16);
        return p;
    };
    auto allocF = [&](size_t n) -> float* {
        off = (off + 255) & ~(size_t)255;
        float* p = (float*)(ws + off);
        off += n * sizeof(float);
        return p;
    };

    // bf16 weights
    __bf16 *w_u1 = allocB(512 * 64),  *w_u2 = allocB(256 * 512), *w_u3 = allocB(128 * 256);
    __bf16 *w_p1 = allocB(512 * 64),  *w_p2 = allocB(256 * 512), *w_p3 = allocB(128 * 256);
    __bf16 *w_t1 = allocB(64 * 32),   *w_t2 = allocB(32 * 64),   *w_t3 = allocB(128 * 32);
    __bf16 *w_a1 = allocB(32 * 128);
    __bf16 *w_T1 = allocB(512 * 1504), *w_T2 = allocB(256 * 512), *w_T3 = allocB(128 * 256);
    __bf16 *w_Tf = allocB(128);
    // bf16 activations
    __bf16 *xu = allocB((size_t)B * 64), *xp = allocB((size_t)B * 64), *xt = allocB((size_t)B * 32);
    __bf16 *h512 = allocB((size_t)B * 512), *h256 = allocB((size_t)B * 256);
    __bf16 *h128 = allocB((size_t)B * 128);
    __bf16 *h64  = allocB((size_t)B * 64),  *h32  = allocB((size_t)B * 32);
    __bf16 *h_att = allocB((size_t)B * 11 * 32);
    __bf16 *all_emb = allocB((size_t)B * 1408);
    __bf16 *bfi = allocB((size_t)B * 128);
    __bf16 *top_in = allocB((size_t)B * 1504);
    // f32 scratch
    float *logits = allocF((size_t)B * 11);
    float *sc_u1 = allocF(512), *sh_u1 = allocF(512);
    float *sc_u2 = allocF(256), *sh_u2 = allocF(256);
    float *sc_u3 = allocF(128), *sh_u3 = allocF(128);
    float *sc_p1 = allocF(512), *sh_p1 = allocF(512);
    float *sc_p2 = allocF(256), *sh_p2 = allocF(256);
    float *sc_p3 = allocF(128), *sh_p3 = allocF(128);
    float *sc_t1 = allocF(64),  *sh_t1 = allocF(64);
    float *sc_t2 = allocF(32),  *sh_t2 = allocF(32);
    float *sc_t3 = allocF(128), *sh_t3 = allocF(128);
    float *sc_a  = allocF(32),  *sh_a  = allocF(32);
    float *sc_T1 = allocF(512), *sh_T1 = allocF(512);
    float *sc_T2 = allocF(256), *sh_T2 = allocF(256);
    float *sc_T3 = allocF(128), *sh_T3 = allocF(128);

    auto conv = [&](int idx, __bf16* dst, int n) {
        k_f32_to_bf16<<<(n + 255) / 256, 256, 0, stream>>>((const float*)d_in[idx], dst, n);
    };
    auto ssl = [&](int ib, int ig, int ibt, int irm, int irv,
                   float* sc, float* sh, int N, int bn) {
        k_scale_shift<<<(N + 255) / 256, 256, 0, stream>>>(
            (const float*)d_in[ib], (const float*)d_in[ig], (const float*)d_in[ibt],
            (const float*)d_in[irm], (const float*)d_in[irv], sc, sh, N, bn);
    };
    auto gemm = [&](const __bf16* X, const __bf16* W, const float* sc, const float* sh,
                    __bf16* Yb, int ldYb, float* Yf, int ldYf,
                    int M, int N, int K, int act) {
        dim3 g(M / BM, (N + BN - 1) / BN);
        k_gemm_bf16<<<g, 256, 0, stream>>>(X, W, sc, sh, Yb, ldYb, Yf, ldYf, M, N, K, act);
    };

    // --- weight / input conversion (every call; deterministic) ---
    conv(0, xu, B * 64); conv(1, xp, B * 64); conv(2, xt, B * 32);
    conv(5, w_u1, 512 * 64);  conv(11, w_u2, 256 * 512); conv(17, w_u3, 128 * 256);
    conv(19, w_p1, 512 * 64); conv(25, w_p2, 256 * 512); conv(31, w_p3, 128 * 256);
    conv(33, w_t1, 64 * 32);  conv(39, w_t2, 32 * 64);   conv(45, w_t3, 128 * 32);
    conv(47, w_a1, 32 * 128);
    k_convert_pad<<<(512 * 1504 + 255) / 256, 256, 0, stream>>>(
        (const float*)d_in[51], w_T1, 512, 1474, 1504);
    conv(57, w_T2, 256 * 512); conv(63, w_T3, 128 * 256); conv(69, w_Tf, 128);

    // --- BN folding ---
    ssl(6, 7, 8, 9, 10,      sc_u1, sh_u1, 512, 1);
    ssl(12, 13, 14, 15, 16,  sc_u2, sh_u2, 256, 1);
    ssl(18, 18, 18, 18, 18,  sc_u3, sh_u3, 128, 0);
    ssl(20, 21, 22, 23, 24,  sc_p1, sh_p1, 512, 1);
    ssl(26, 27, 28, 29, 30,  sc_p2, sh_p2, 256, 1);
    ssl(32, 32, 32, 32, 32,  sc_p3, sh_p3, 128, 0);
    ssl(34, 35, 36, 37, 38,  sc_t1, sh_t1, 64, 1);
    ssl(40, 41, 42, 43, 44,  sc_t2, sh_t2, 32, 1);
    ssl(46, 46, 46, 46, 46,  sc_t3, sh_t3, 128, 0);
    ssl(48, 48, 48, 48, 48,  sc_a,  sh_a,  32, 0);
    ssl(52, 53, 54, 55, 56,  sc_T1, sh_T1, 512, 1);
    ssl(58, 59, 60, 61, 62,  sc_T2, sh_T2, 256, 1);
    ssl(64, 65, 66, 67, 68,  sc_T3, sh_T3, 128, 1);

    const size_t bot0 = (size_t)12 * B;  // d_out offset of user_bottom

    // --- bottom MLPs (WMMA) ---
    gemm(xu, w_u1, sc_u1, sh_u1, h512, 512, nullptr, 0, B, 512, 64, 1);
    gemm(h512, w_u2, sc_u2, sh_u2, h256, 256, nullptr, 0, B, 256, 512, 1);
    gemm(h256, w_u3, sc_u3, sh_u3, all_emb + 0 * 128, 1408, fo + bot0, 128, B, 128, 256, 0);

    gemm(xp, w_p1, sc_p1, sh_p1, h512, 512, nullptr, 0, B, 512, 64, 1);
    gemm(h512, w_p2, sc_p2, sh_p2, h256, 256, nullptr, 0, B, 256, 512, 1);
    gemm(h256, w_p3, sc_p3, sh_p3, all_emb + 1 * 128, 1408,
         fo + bot0 + (size_t)B * 128, 128, B, 128, 256, 0);

    gemm(xt, w_t1, sc_t1, sh_t1, h64, 64, nullptr, 0, B, 64, 32, 1);
    gemm(h64, w_t2, sc_t2, sh_t2, h32, 32, nullptr, 0, B, 32, 64, 1);
    gemm(h32, w_t3, sc_t3, sh_t3, all_emb + 2 * 128, 1408,
         fo + bot0 + (size_t)2 * B * 128, 128, B, 128, 32, 0);

    // --- embeddings ---
    k_gather<<<(B * 1024 + 255) / 256, 256, 0, stream>>>(
        (const float*)d_in[3], (const float*)d_in[4],
        (const int*)d_in[71], (const int*)d_in[72], all_emb, B, vocab);

    // --- attention: h = tanh(all_emb @ Wa1^T + ba1) as (B*11)x32x128 WMMA GEMM ---
    gemm(all_emb, w_a1, sc_a, sh_a, h_att, 32, nullptr, 0, B * 11, 32, 128, 2);
    k_logits<<<(B * 11 + 255) / 256, 256, 0, stream>>>(
        h_att, (const float*)d_in[49], (const float*)d_in[50], logits, B * 11);
    k_softmax11<<<(B + 255) / 256, 256, 0, stream>>>(logits, fo + B, B);
    k_bfi<<<(B * 128 + 255) / 256, 256, 0, stream>>>(fo + B, all_emb, bfi, B);

    // --- top input assembly ---
    k_flat<<<(B * 1408 + 255) / 256, 256, 0, stream>>>(all_emb, top_in, B);
    k_inter<<<B, 128, 0, stream>>>(bfi, all_emb, top_in, B);

    // --- top MLP (WMMA) ---
    gemm(top_in, w_T1, sc_T1, sh_T1, h512, 512, nullptr, 0, B, 512, 1504, 1);
    gemm(h512, w_T2, sc_T2, sh_T2, h256, 256, nullptr, 0, B, 256, 512, 1);
    gemm(h256, w_T3, sc_T3, sh_T3, h128, 128, nullptr, 0, B, 128, 256, 1);
    k_final<<<(B + 255) / 256, 256, 0, stream>>>(
        h128, w_Tf, (const float*)d_in[70], fo, B);
}
